// MTAFN_5016521801952
// MI455X (gfx1250) — compile-verified, matrix operand order checked by reading, not run on hardware
//
#include <hip/hip_runtime.h>

// ---------------------------------------------------------------- types
typedef __attribute__((ext_vector_type(16))) _Float16 v16h;
typedef __attribute__((ext_vector_type(8)))  float    v8f;

#define WMMA_F32_F16(a,b,c) \
  __builtin_amdgcn_wmma_f32_16x16x32_f16(false,(a),false,(b),(short)0,(c),false,false)

__device__ __forceinline__ int pat(int lane, int i) {
  int kb = (lane >> 4) * 8;
  return (i < 8) ? (kb + i) : (16 + kb + (i - 8));
}
// fast transcendental paths: v_exp_f32 + v_rcp_f32 (no IEEE divide chain) --
// these sit on the LSTM's serial critical path (1024 sequential steps).
__device__ __forceinline__ float fsigmoid(float x) {
  return __builtin_amdgcn_rcpf(1.f + __expf(-x));
}
__device__ __forceinline__ float ftanh(float x) {
  x = fminf(fmaxf(x, -15.f), 15.f);
  float e = __expf(-2.f*x);
  return (1.f - e) * __builtin_amdgcn_rcpf(1.f + e);
}

// ---------------------------------------------------------------- consts
#define Bsz 16
#define CIN 3
#define LEN 1024
#define HID 256
#define C2 128
#define NHd 8
#define DK 32

// db4 taps already reversed for cross-correlation (conv1d w = DB4[::-1])
__constant__ float DB4H[8] = {-0.230377813308896f, 0.714846570552542f, -0.63088076792959f,
                               0.027983769416984f, 0.187034811718881f, 0.030841381835987f,
                              -0.032883011666983f,-0.010597401784997f};
__constant__ float DB4G[8] = { 0.010597401784997f,-0.032883011666983f,-0.030841381835987f,
                               0.187034811718881f,-0.027983769416984f,-0.63088076792959f,
                              -0.714846570552542f,-0.230377813308896f};

// ---------------------------------------------------------------- small kernels
__global__ void wavelet_kernel(const float* __restrict__ x, float* __restrict__ ap,
                               float* __restrict__ de) {
  int idx = blockIdx.x*blockDim.x + threadIdx.x;
  if (idx >= Bsz*CIN*512) return;
  int i = idx % 512, c = (idx/512)%CIN, b = idx/(512*CIN);
  float sa = 0.f, sd = 0.f;
  for (int t = 0; t < 8; ++t) {
    int li = 2*i + t - 3;
    if (li < 0 || li >= LEN) continue;
    float v = x[((size_t)(b*CIN)+c)*LEN + li];
    sa += v*DB4H[t]; sd += v*DB4G[t];
  }
  ap[((size_t)(b*CIN)+c)*512 + i] = sa;
  de[((size_t)(b*CIN)+c)*512 + i] = sd;
}

__global__ void dwconv_kernel(const float* __restrict__ in, const float* __restrict__ w,
                              const float* __restrict__ bias, float* __restrict__ out,
                              int C, int L, int K, int padv) {
  int idx = blockIdx.x*blockDim.x + threadIdx.x;
  if (idx >= Bsz*C*L) return;
  int l = idx % L, c = (idx/L)%C, b = idx/(L*C);
  float s = bias[c];
  for (int t = 0; t < K; ++t) {
    int li = l + t - padv;
    if (li < 0 || li >= L) continue;
    s += in[((size_t)(b*C)+c)*L + li] * w[c*K + t];
  }
  out[((size_t)(b*C)+c)*L + l] = s;
}

// dense conv1d, stride 1; act: 0 none, 1 relu, 2 sigmoid; in_blc: input layout (B,L,Ci)
__global__ void conv1d_kernel(const float* __restrict__ in, const float* __restrict__ w,
                              const float* __restrict__ bias, float* __restrict__ out,
                              int Ci, int Co, int L, int K, int padv,
                              int coff, int Ctot, int act, int in_blc) {
  int idx = blockIdx.x*blockDim.x + threadIdx.x;
  if (idx >= Bsz*Co*L) return;
  int l = idx % L, o = (idx/L)%Co, b = idx/(L*Co);
  float s = bias[o];
  for (int c = 0; c < Ci; ++c) {
    const float* wp = w + (size_t)(o*Ci + c)*K;
    for (int t = 0; t < K; ++t) {
      int li = l + t - padv;
      if (li < 0 || li >= L) continue;
      float xv = in_blc ? in[((size_t)b*L + li)*Ci + c]
                        : in[((size_t)(b*Ci)+c)*L + li];
      s += xv * wp[t];
    }
  }
  if (act == 1) s = fmaxf(s, 0.f);
  else if (act == 2) s = fsigmoid(s);
  out[((size_t)(b*Ctot) + coff + o)*L + l] = s;
}

__global__ void bn_stats_kernel(const float* __restrict__ in, float* __restrict__ stats,
                                int C, int L) {
  int c = blockIdx.x;
  __shared__ float s1[256], s2[256];
  float a = 0.f, q = 0.f;
  int N = Bsz*L;
  for (int i = threadIdx.x; i < N; i += 256) {
    int b = i / L, l = i % L;
    float v = in[((size_t)(b*C)+c)*L + l];
    a += v; q += v*v;
  }
  s1[threadIdx.x] = a; s2[threadIdx.x] = q; __syncthreads();
  for (int st = 128; st > 0; st >>= 1) {
    if (threadIdx.x < st) { s1[threadIdx.x]+=s1[threadIdx.x+st]; s2[threadIdx.x]+=s2[threadIdx.x+st]; }
    __syncthreads();
  }
  if (threadIdx.x == 0) {
    float m = s1[0]/N; float v = s2[0]/N - m*m;
    stats[2*c] = m; stats[2*c+1] = v;
  }
}

__global__ void bn_apply_kernel(const float* __restrict__ in, float* __restrict__ out,
                                const float* __restrict__ stats, const float* __restrict__ g,
                                const float* __restrict__ bt, int C, int L, int relu) {
  int idx = blockIdx.x*blockDim.x + threadIdx.x;
  if (idx >= Bsz*C*L) return;
  int c = (idx/L)%C;
  float m = stats[2*c], v = stats[2*c+1];
  float y = (in[idx] - m)*rsqrtf(v + 1e-5f)*g[c] + bt[c];
  if (relu) y = fmaxf(y, 0.f);
  out[idx] = y;
}

__global__ void gap_kernel(const float* __restrict__ in, float* __restrict__ out, int C, int L) {
  int idx = blockIdx.x*blockDim.x + threadIdx.x;
  if (idx >= Bsz*C) return;
  int c = idx % C, b = idx / C;
  float s = 0.f;
  for (int l = 0; l < L; ++l) s += in[((size_t)(b*C)+c)*L + l];
  out[idx] = s / L;
}

__global__ void ca1_kernel(const float* __restrict__ gp, const float* __restrict__ w,
                           const float* __restrict__ bias, float* __restrict__ mid,
                           int C, int Cq) {
  int idx = blockIdx.x*blockDim.x + threadIdx.x;
  if (idx >= Bsz*Cq) return;
  int o = idx % Cq, b = idx / Cq;
  float s = bias[o];
  for (int c = 0; c < C; ++c) s += gp[b*C + c]*w[o*C + c];
  mid[idx] = fmaxf(s, 0.f);
}

__global__ void ca2_kernel(const float* __restrict__ mid, const float* __restrict__ w,
                           const float* __restrict__ bias, float* __restrict__ ca,
                           int C, int Cq) {
  int idx = blockIdx.x*blockDim.x + threadIdx.x;
  if (idx >= Bsz*C) return;
  int c = idx % C, b = idx / C;
  float s = bias[c];
  for (int o = 0; o < Cq; ++o) s += mid[b*Cq + o]*w[c*Cq + o];
  ca[idx] = fsigmoid(s);
}

__global__ void cr_kernel(float* __restrict__ ms, const float* __restrict__ ca,
                          float* __restrict__ cr, int C, int L) {
  int idx = blockIdx.x*blockDim.x + threadIdx.x;
  if (idx >= Bsz*C*L) return;
  int c = (idx/L)%C, b = idx/(L*C);
  cr[idx] = ms[idx]*ca[b*C + c];
}

__global__ void sp_kernel(const float* __restrict__ cr, float* __restrict__ sp, int C, int L) {
  int idx = blockIdx.x*blockDim.x + threadIdx.x;
  if (idx >= Bsz*L) return;
  int l = idx % L, b = idx / L;
  float s = 0.f, mx = -3.0e38f;
  for (int c = 0; c < C; ++c) {
    float v = cr[((size_t)(b*C)+c)*L + l];
    s += v; mx = fmaxf(mx, v);
  }
  sp[((size_t)(b*2)+0)*L + l] = s / C;
  sp[((size_t)(b*2)+1)*L + l] = mx;
}

__global__ void sa_kernel(const float* __restrict__ sp, const float* __restrict__ w,
                          const float* __restrict__ bias, float* __restrict__ sa, int L) {
  int idx = blockIdx.x*blockDim.x + threadIdx.x;
  if (idx >= Bsz*L) return;
  int l = idx % L, b = idx / L;
  float s = bias[0];
  for (int c = 0; c < 2; ++c)
    for (int t = 0; t < 7; ++t) {
      int li = l + t - 3;
      if (li < 0 || li >= L) continue;
      s += sp[((size_t)(b*2)+c)*L + li]*w[c*7 + t];
    }
  sa[(size_t)b*L + l] = fsigmoid(s);
}

__global__ void res_kernel(float* __restrict__ cr, const float* __restrict__ sa,
                           const float* __restrict__ xin, int C, int L) {
  int idx = blockIdx.x*blockDim.x + threadIdx.x;
  if (idx >= Bsz*C*L) return;
  int l = idx % L, b = idx/(L*C);
  cr[idx] = cr[idx]*sa[(size_t)b*L + l] + xin[idx];
}

__global__ void transpose_xs_kernel(const float* __restrict__ t, float* __restrict__ xs) {
  int idx = blockIdx.x*blockDim.x + threadIdx.x;
  if (idx >= LEN*Bsz*C2) return;
  int c = idx % C2, b = (idx/C2)%Bsz, l = idx/(C2*Bsz);
  xs[idx] = t[((size_t)(b*C2)+c)*LEN + l];
}

__global__ void timefeat_kernel(const float* __restrict__ t, const float* __restrict__ hf,
                                const float* __restrict__ hb, float* __restrict__ tf) {
  int idx = blockIdx.x*blockDim.x + threadIdx.x;
  if (idx >= Bsz*LEN*HID) return;
  int c = idx % HID, l = (idx/HID)%LEN, b = idx/(HID*LEN);
  float v;
  if (c < C2)      v = t[((size_t)(b*C2)+c)*LEN + l];
  else if (c < 192) v = hf[((size_t)(l*Bsz)+b)*64 + (c-128)];
  else              v = hb[((size_t)(l*Bsz)+b)*64 + (c-192)];
  tf[idx] = v;
}

__global__ void interp_kernel(const float* __restrict__ in, float* __restrict__ out,
                              int C, int Lin, int Lout) {
  int idx = blockIdx.x*blockDim.x + threadIdx.x;
  if (idx >= Bsz*C*Lout) return;
  int j = idx % Lout, c = (idx/Lout)%C, b = idx/(Lout*C);
  float src = (j + 0.5f)*((float)Lin/(float)Lout) - 0.5f;
  src = fminf(fmaxf(src, 0.f), (float)(Lin-1));
  int lo = (int)floorf(src);
  int hi = lo + 1; if (hi > Lin-1) hi = Lin-1;
  float w = src - (float)lo;
  const float* p = in + ((size_t)(b*C)+c)*Lin;
  out[idx] = p[lo]*(1.f-w) + p[hi]*w;
}

__global__ void freq_build_kernel(const float* __restrict__ fa, const float* __restrict__ fd,
                                  float* __restrict__ fq) {
  int idx = blockIdx.x*blockDim.x + threadIdx.x;
  if (idx >= Bsz*LEN*HID) return;
  int c = idx % HID, l = (idx/HID)%LEN, b = idx/(HID*LEN);
  float v = 0.f;
  if (c < 64)       v = fa[((size_t)(b*64)+c)*LEN + l];
  else if (c < 128) v = fd[((size_t)(b*64)+(c-64))*LEN + l];
  fq[idx] = v;
}

__global__ void resln_kernel(const float* __restrict__ op, const float* __restrict__ tf,
                             const float* __restrict__ g, const float* __restrict__ bt,
                             float* __restrict__ out) {
  int row = blockIdx.x*blockDim.x + threadIdx.x;
  if (row >= Bsz*LEN) return;
  const float* a = op + (size_t)row*HID;
  const float* t = tf + (size_t)row*HID;
  float m = 0.f;
  for (int i = 0; i < HID; ++i) m += a[i] + t[i];
  m *= (1.f/HID);
  float v = 0.f;
  for (int i = 0; i < HID; ++i) { float d = a[i]+t[i]-m; v += d*d; }
  v *= (1.f/HID);
  float inv = rsqrtf(v + 1e-5f);
  float* o = out + (size_t)row*HID;
  for (int i = 0; i < HID; ++i) o[i] = (a[i]+t[i]-m)*inv*g[i] + bt[i];
}

__global__ void pool_kernel(const float* __restrict__ fused, float* __restrict__ pooled) {
  int idx = blockIdx.x*blockDim.x + threadIdx.x;
  if (idx >= Bsz*HID) return;
  int c = idx % HID, b = idx / HID;
  float s = 0.f;
  for (int l = 0; l < LEN; ++l) s += fused[((size_t)(b*LEN)+l)*HID + c];
  pooled[idx] = s / LEN;
}

__global__ void linear_kernel(const float* __restrict__ in, const float* __restrict__ w,
                              const float* __restrict__ bias, float* __restrict__ out,
                              int Kin, int Nout, int act) {
  int idx = blockIdx.x*blockDim.x + threadIdx.x;
  if (idx >= Bsz*Nout) return;
  int n = idx % Nout, b = idx / Nout;
  float s = bias[n];
  for (int k = 0; k < Kin; ++k) s += in[(size_t)b*Kin + k]*w[(size_t)n*Kin + k];
  if (act == 1) s = fmaxf(s, 0.f);
  else if (act == 2) s = fsigmoid(s);
  out[idx] = s;
}

__global__ void task_kernel(const float* __restrict__ w, float* __restrict__ out) {
  if (threadIdx.x | blockIdx.x) return;
  float m = fmaxf(w[0], fmaxf(w[1], w[2]));
  float e0 = __expf(w[0]-m), e1 = __expf(w[1]-m), e2 = __expf(w[2]-m);
  float s = e0+e1+e2;
  out[0]=e0/s; out[1]=e1/s; out[2]=e2/s;
}

// ---------------------------------------------------------------- WMMA GEMM
// C[M,N] = A[M,K] @ W[N,K]^T (+bias).  M%128==0, N%64==0, K%32==0.
template<int OUTF16>
__global__ __launch_bounds__(256) void gemm_kernel(
    const float* __restrict__ A, const float* __restrict__ Wt,
    const float* __restrict__ bias, void* __restrict__ Cp, int M, int N, int K) {
  __shared__ _Float16 As[128][40];
  __shared__ _Float16 Bs2[32][72];
  int tid = threadIdx.x, lane = tid & 31, wave = tid >> 5;
  int m0 = blockIdx.x * 128, n0 = blockIdx.y * 64;
  v8f acc[4] = {{},{},{},{}};
  for (int k0 = 0; k0 < K; k0 += 32) {
    {
      int row = tid >> 1, cs = (tid & 1)*16;
      const float* apx = A + (size_t)(m0 + row)*K + k0 + cs;
      for (int i = 0; i < 16; ++i) As[row][cs+i] = (_Float16)apx[i];
      int n = tid & 63, ks = (tid >> 6)*8;
      const float* wp = Wt + (size_t)(n0 + n)*K + k0 + ks;
      for (int i = 0; i < 8; ++i) Bs2[ks+i][n] = (_Float16)wp[i];
    }
    __syncthreads();
    v16h a;
    { int r = lane & 15;
      for (int i = 0; i < 16; ++i) a[i] = As[wave*16 + r][pat(lane,i)]; }
    for (int j = 0; j < 4; ++j) {
      v16h bfr;
      int c = j*16 + (lane & 15);
      for (int i = 0; i < 16; ++i) bfr[i] = Bs2[pat(lane,i)][c];
      acc[j] = WMMA_F32_F16(a, bfr, acc[j]);
    }
    __syncthreads();
  }
  for (int j = 0; j < 4; ++j)
    for (int r = 0; r < 8; ++r) {
      int m = m0 + wave*16 + (lane>>4)*8 + r;
      int n = n0 + j*16 + (lane & 15);
      float v = acc[j][r] + (bias ? bias[n] : 0.f);
      if (OUTF16) ((_Float16*)Cp)[(size_t)m*N + n] = (_Float16)v;
      else        ((float*)Cp)[(size_t)m*N + n] = v;
    }
}

// ---------------------------------------------------------------- LSTM (persistent, WMMA recurrence)
__global__ __launch_bounds__(512) void lstm_kernel(
    const float* __restrict__ ihF, const float* __restrict__ ihB,
    const float* __restrict__ whhF, const float* __restrict__ whhB,
    const float* __restrict__ bihF, const float* __restrict__ bhhF,
    const float* __restrict__ bihB, const float* __restrict__ bhhB,
    float* __restrict__ outF, float* __restrict__ outB) {
  const int dir = blockIdx.x;
  const float* ih  = dir ? ihB  : ihF;
  const float* whh = dir ? whhB : whhF;
  const float* bih = dir ? bihB : bihF;
  const float* bhh = dir ? bhhB : bhhF;
  float* outp      = dir ? outB : outF;

  __shared__ _Float16 hs16[16][72];
  __shared__ float cst[16][64];
  __shared__ float zs[16][260];
  __shared__ float bsum[256];

  int tid = threadIdx.x, lane = tid & 31, wave = tid >> 5;
  for (int i = tid; i < 16*64; i += 512) {
    int b = i>>6, u = i&63;
    cst[b][u] = 0.f; hs16[b][u] = (_Float16)0.f;
  }
  for (int j = tid; j < 256; j += 512) bsum[j] = bih[j] + bhh[j];
  __syncthreads();

  // preload Whh^T B-fragments for this wave's 16 output columns (resident all 1024 steps)
  v16h bf0, bf1;
  { int n = lane & 15;
    for (int i = 0; i < 16; ++i) {
      int k = pat(lane, i);
      bf0[i] = (_Float16)whh[(size_t)(wave*16+n)*64 + k];
      bf1[i] = (_Float16)whh[(size_t)(wave*16+n)*64 + 32 + k];
    } }

  for (int s = 0; s < LEN; ++s) {
    int t = dir ? (LEN-1-s) : s;
    v16h a0, a1;
    { int r = lane & 15;
      for (int i = 0; i < 16; ++i) {
        int k = pat(lane, i);
        a0[i] = hs16[r][k];
        a1[i] = hs16[r][32 + k];
      } }
    v8f acc = {};
    acc = WMMA_F32_F16(a0, bf0, acc);
    acc = WMMA_F32_F16(a1, bf1, acc);
    { int col = wave*16 + (lane & 15);
      for (int q = 0; q < 8; ++q) zs[(lane>>4)*8 + q][col] = acc[q]; }
    __syncthreads();
    for (int e = tid; e < 1024; e += 512) {
      int b = e >> 6, u = e & 63;
      const float* ihp = ih + ((size_t)t*16 + b)*256;
      float zi = zs[b][u]      + ihp[u]      + bsum[u];
      float zf = zs[b][64+u]   + ihp[64+u]   + bsum[64+u];
      float zg = zs[b][128+u]  + ihp[128+u]  + bsum[128+u];
      float zo = zs[b][192+u]  + ihp[192+u]  + bsum[192+u];
      float cv = fsigmoid(zf)*cst[b][u] + fsigmoid(zi)*ftanh(zg);
      float hv = fsigmoid(zo)*ftanh(cv);
      cst[b][u] = cv;
      outp[((size_t)t*16 + b)*64 + u] = hv;
      hs16[b][u] = (_Float16)hv;
    }
    __syncthreads();
  }
}

// ---------------------------------------------------------------- flash attention (1 wave / 16-query tile)
__global__ __launch_bounds__(32) void flash_kernel(
    const _Float16* __restrict__ qh, const _Float16* __restrict__ kh,
    const _Float16* __restrict__ vh, float* __restrict__ ctx) {
  int blk = blockIdx.x;
  int qt = blk & 63, h = (blk >> 6) & 7, b = blk >> 9;
  int lane = threadIdx.x;
  int q0 = qt * 16;
  const float sc = 0.1767766952966369f;  // 1/sqrt(32)
  __shared__ _Float16 ps[16][40];

  v16h qa;
  { int r = lane & 15;
    const _Float16* qp = qh + (((size_t)b*LEN + q0 + r)*HID) + h*DK;
    for (int i = 0; i < 16; ++i) qa[i] = qp[pat(lane,i)]; }

  v8f o0 = {}, o1 = {};
  float mrow[8], lrow[8];
  for (int r = 0; r < 8; ++r) { mrow[r] = -3.0e38f; lrow[r] = 0.f; }

  for (int kt = 0; kt < 32; ++kt) {
    int kb = kt*32;
    v16h bk0, bk1;
    { int n = lane & 15;
      const _Float16* kp0 = kh + (((size_t)b*LEN + kb + n)*HID) + h*DK;
      const _Float16* kp1 = kh + (((size_t)b*LEN + kb + 16 + n)*HID) + h*DK;
      for (int i = 0; i < 16; ++i) { int d = pat(lane,i); bk0[i]=kp0[d]; bk1[i]=kp1[d]; } }
    v8f z = {};
    v8f s0 = WMMA_F32_F16(qa, bk0, z);
    v8f s1 = WMMA_F32_F16(qa, bk1, z);
    for (int r = 0; r < 8; ++r) {
      float v0 = sc*s0[r], v1 = sc*s1[r];
      float mx = fmaxf(v0, v1);
      for (int m = 1; m < 16; m <<= 1) mx = fmaxf(mx, __shfl_xor(mx, m, 32));
      float mnew = fmaxf(mrow[r], mx);
      float alpha = __expf(mrow[r] - mnew);
      float p0 = __expf(v0 - mnew), p1 = __expf(v1 - mnew);
      float psum = p0 + p1;
      for (int m = 1; m < 16; m <<= 1) psum += __shfl_xor(psum, m, 32);
      lrow[r] = lrow[r]*alpha + psum;
      mrow[r] = mnew;
      o0[r] *= alpha; o1[r] *= alpha;
      int mr = (lane>>4)*8 + r;
      ps[mr][lane & 15] = (_Float16)p0;
      ps[mr][16 + (lane & 15)] = (_Float16)p1;
    }
    __syncthreads();  // single wave -> S_NOP; LDS kept in-order per wave
    v16h pa;
    { int r = lane & 15;
      for (int i = 0; i < 16; ++i) pa[i] = ps[r][pat(lane,i)]; }
    v16h bv0, bv1;
    { int n = lane & 15;
      for (int i = 0; i < 16; ++i) {
        int kk = pat(lane,i);
        const _Float16* vp = vh + (((size_t)b*LEN + kb + kk)*HID) + h*DK;
        bv0[i] = vp[n];
        bv1[i] = vp[16 + n];
      } }
    o0 = WMMA_F32_F16(pa, bv0, o0);
    o1 = WMMA_F32_F16(pa, bv1, o1);
    __syncthreads();
  }
  { int n = lane & 15;
    for (int r = 0; r < 8; ++r) {
      int m = (lane>>4)*8 + r;
      float inv = __builtin_amdgcn_rcpf(lrow[r]);
      size_t base = (((size_t)b*LEN) + q0 + m)*HID + h*DK;
      ctx[base + n]      = o0[r]*inv;
      ctx[base + 16 + n] = o1[r]*inv;
    } }
}

// ---------------------------------------------------------------- input index map (dict insertion order)
enum {
  IN_X = 0, IN_DW_W, IN_DW_B, IN_PW_W, IN_PW_B, IN_BN1_G, IN_BN1_B,
  IN_MSA_T,                      // 7  (16 arrays)
  IN_LSTMF = IN_MSA_T + 16,      // 23 w_ih, w_hh, b_ih, b_hh
  IN_LSTMB = IN_LSTMF + 4,       // 27
  IN_FA_W  = IN_LSTMB + 4,       // 31
  IN_FA_B, IN_FA_BN_G, IN_FA_BN_B,
  IN_MSA_FA,                     // 35
  IN_FD_W = IN_MSA_FA + 16,      // 51
  IN_FD_B, IN_FD_BN_G, IN_FD_BN_B,
  IN_MSA_FD,                     // 55
  IN_WQ = IN_MSA_FD + 16,        // 71
  IN_BQ, IN_WK, IN_BK, IN_WV, IN_BV, IN_WO, IN_BO,
  IN_LN_G, IN_LN_B,
  IN_CLS1_W, IN_CLS1_B, IN_CLS2_W, IN_CLS2_B,
  IN_LOC1_W, IN_LOC1_B, IN_LOC2_W, IN_LOC2_B, IN_LOC3_W, IN_LOC3_B,
  IN_SEV1_W, IN_SEV1_B, IN_SEV2_W, IN_SEV2_B,
  IN_TASK_W
};

struct MsaBufs { float *ms, *cr, *gap, *mid, *ca, *sp, *sa, *stats; };

static void run_msa(const float* xin, float* outb, void* const* d_in, int pb,
                    int C, int Lc, const MsaBufs& S, hipStream_t st) {
  int Cq = C/4;
  const int ks[4] = {1,3,5,7};
  for (int i = 0; i < 4; ++i) {
    int tot = Bsz*Cq*Lc;
    conv1d_kernel<<<(tot+255)/256,256,0,st>>>(xin, (const float*)d_in[pb+2*i],
        (const float*)d_in[pb+2*i+1], S.ms, C, Cq, Lc, ks[i], ks[i]/2, i*Cq, C, 0, 0);
  }
  gap_kernel<<<(Bsz*C+255)/256,256,0,st>>>(S.ms, S.gap, C, Lc);
  ca1_kernel<<<(Bsz*Cq+255)/256,256,0,st>>>(S.gap, (const float*)d_in[pb+8],
      (const float*)d_in[pb+9], S.mid, C, Cq);
  ca2_kernel<<<(Bsz*C+255)/256,256,0,st>>>(S.mid, (const float*)d_in[pb+10],
      (const float*)d_in[pb+11], S.ca, C, Cq);
  int tot = Bsz*C*Lc;
  cr_kernel<<<(tot+255)/256,256,0,st>>>(S.ms, S.ca, S.cr, C, Lc);
  sp_kernel<<<(Bsz*Lc+255)/256,256,0,st>>>(S.cr, S.sp, C, Lc);
  sa_kernel<<<(Bsz*Lc+255)/256,256,0,st>>>(S.sp, (const float*)d_in[pb+12],
      (const float*)d_in[pb+13], S.sa, Lc);
  res_kernel<<<(tot+255)/256,256,0,st>>>(S.cr, S.sa, xin, C, Lc);
  bn_stats_kernel<<<C,256,0,st>>>(S.cr, S.stats, C, Lc);
  bn_apply_kernel<<<(tot+255)/256,256,0,st>>>(S.cr, outb, S.stats,
      (const float*)d_in[pb+14], (const float*)d_in[pb+15], C, Lc, 0);
}

// ---------------------------------------------------------------- launch
extern "C" void kernel_launch(void* const* d_in, const int* in_sizes, int n_in,
                              void* d_out, int out_size, void* d_ws, size_t ws_size,
                              hipStream_t stream) {
  (void)in_sizes; (void)n_in; (void)out_size; (void)ws_size;
  const float* x = (const float*)d_in[IN_X];
  float* Wsp = (float*)d_ws;
  size_t off = 0;
  auto A = [&](size_t n) { float* p = Wsp + off; off += n; return p; };

  float* approx = A(16*3*512);
  float* detail = A(16*3*512);
  float* tdw    = A(16*3*1024);
  float* tpw    = A((size_t)16*128*1024);
  float* msat   = A((size_t)16*128*1024);
  float* ms_s   = A((size_t)16*128*1024);
  float* cr_s   = A((size_t)16*128*1024);
  float* gap_s  = A(16*128);
  float* mid_s  = A(16*32);
  float* ca_s   = A(16*128);
  float* sp_s   = A(16*2*1024);
  float* sa_s   = A(16*1024);
  float* stats  = A(2*256);
  float* conv_s = A((size_t)16*64*512);
  float* msas   = A((size_t)16*64*512);
  float* fa_i   = A((size_t)16*64*1024);
  float* fd_i   = A((size_t)16*64*1024);
  float* xsb    = A((size_t)1024*16*128);
  float* ih_f   = A((size_t)1024*16*256);
  float* ih_b   = A((size_t)1024*16*256);
  float* h_f    = A((size_t)1024*16*64);
  float* h_b    = A((size_t)1024*16*64);
  float* tf     = A((size_t)16*1024*256);
  float* freq   = A((size_t)16*1024*256);
  _Float16* qh  = (_Float16*)A((size_t)16*1024*128);
  _Float16* kh  = (_Float16*)A((size_t)16*1024*128);
  _Float16* vh  = (_Float16*)A((size_t)16*1024*128);
  float* fused  = A((size_t)16*1024*256);
  float* loc1b  = A((size_t)16*128*1024);
  float* loc2b  = A((size_t)16*64*1024);
  float* pooled = A(16*256);
  float* mid1   = A(16*128);
  float* mid2   = A(16*64);
  float* ctx    = ih_f;   // reuse: ih buffers dead after LSTM
  float* oproj  = ih_b;

  MsaBufs S = { ms_s, cr_s, gap_s, mid_s, ca_s, sp_s, sa_s, stats };
  float* outp = (float*)d_out;
  float* out_cls = outp;             // 16*5
  float* out_loc = outp + 80;        // 16*1024
  float* out_sev = outp + 80+16384;  // 16
  float* out_tw  = outp + 80+16384+16;

  // 1) wavelet split
  wavelet_kernel<<<(Bsz*CIN*512+255)/256,256,0,stream>>>(x, approx, detail);
  // 2) time path: dw k7 -> pw -> BN+relu -> MSA
  dwconv_kernel<<<(Bsz*CIN*LEN+255)/256,256,0,stream>>>(x, (const float*)d_in[IN_DW_W],
      (const float*)d_in[IN_DW_B], tdw, CIN, LEN, 7, 3);
  conv1d_kernel<<<(Bsz*C2*LEN+255)/256,256,0,stream>>>(tdw, (const float*)d_in[IN_PW_W],
      (const float*)d_in[IN_PW_B], tpw, CIN, C2, LEN, 1, 0, 0, C2, 0, 0);
  bn_stats_kernel<<<C2,256,0,stream>>>(tpw, stats, C2, LEN);
  bn_apply_kernel<<<(Bsz*C2*LEN+255)/256,256,0,stream>>>(tpw, tpw, stats,
      (const float*)d_in[IN_BN1_G], (const float*)d_in[IN_BN1_B], C2, LEN, 1);
  run_msa(tpw, msat, d_in, IN_MSA_T, C2, LEN, S, stream);
  // 3) LSTM: input projections as big WMMA GEMMs, then persistent recurrence
  transpose_xs_kernel<<<(LEN*Bsz*C2+255)/256,256,0,stream>>>(msat, xsb);
  {
    dim3 g(16384/128, 256/64);
    gemm_kernel<0><<<g,256,0,stream>>>(xsb, (const float*)d_in[IN_LSTMF+0], nullptr,
                                       ih_f, 16384, 256, 128);
    gemm_kernel<0><<<g,256,0,stream>>>(xsb, (const float*)d_in[IN_LSTMB+0], nullptr,
                                       ih_b, 16384, 256, 128);
  }
  lstm_kernel<<<2,512,0,stream>>>(ih_f, ih_b,
      (const float*)d_in[IN_LSTMF+1], (const float*)d_in[IN_LSTMB+1],
      (const float*)d_in[IN_LSTMF+2], (const float*)d_in[IN_LSTMF+3],
      (const float*)d_in[IN_LSTMB+2], (const float*)d_in[IN_LSTMB+3],
      h_f, h_b);
  // 4) frequency paths (approx then detail; reuse scratch)
  conv1d_kernel<<<(Bsz*64*512+255)/256,256,0,stream>>>(approx, (const float*)d_in[IN_FA_W],
      (const float*)d_in[IN_FA_B], conv_s, CIN, 64, 512, 5, 2, 0, 64, 0, 0);
  bn_stats_kernel<<<64,256,0,stream>>>(conv_s, stats, 64, 512);
  bn_apply_kernel<<<(Bsz*64*512+255)/256,256,0,stream>>>(conv_s, conv_s, stats,
      (const float*)d_in[IN_FA_BN_G], (const float*)d_in[IN_FA_BN_B], 64, 512, 1);
  run_msa(conv_s, msas, d_in, IN_MSA_FA, 64, 512, S, stream);
  interp_kernel<<<(Bsz*64*LEN+255)/256,256,0,stream>>>(msas, fa_i, 64, 512, LEN);
  conv1d_kernel<<<(Bsz*64*512+255)/256,256,0,stream>>>(detail, (const float*)d_in[IN_FD_W],
      (const float*)d_in[IN_FD_B], conv_s, CIN, 64, 512, 3, 1, 0, 64, 0, 0);
  bn_stats_kernel<<<64,256,0,stream>>>(conv_s, stats, 64, 512);
  bn_apply_kernel<<<(Bsz*64*512+255)/256,256,0,stream>>>(conv_s, conv_s, stats,
      (const float*)d_in[IN_FD_BN_G], (const float*)d_in[IN_FD_BN_B], 64, 512, 1);
  run_msa(conv_s, msas, d_in, IN_MSA_FD, 64, 512, S, stream);
  interp_kernel<<<(Bsz*64*LEN+255)/256,256,0,stream>>>(msas, fd_i, 64, 512, LEN);
  // 5) feature assembly
  timefeat_kernel<<<(Bsz*LEN*HID+255)/256,256,0,stream>>>(msat, h_f, h_b, tf);
  freq_build_kernel<<<(Bsz*LEN*HID+255)/256,256,0,stream>>>(fa_i, fd_i, freq);
  // 6) QKV projections (f16 out for flash) + flash attention + output proj
  {
    dim3 g(16384/128, 256/64);
    gemm_kernel<1><<<g,256,0,stream>>>(tf,   (const float*)d_in[IN_WQ],
        (const float*)d_in[IN_BQ], qh, 16384, 256, 256);
    gemm_kernel<1><<<g,256,0,stream>>>(freq, (const float*)d_in[IN_WK],
        (const float*)d_in[IN_BK], kh, 16384, 256, 256);
    gemm_kernel<1><<<g,256,0,stream>>>(freq, (const float*)d_in[IN_WV],
        (const float*)d_in[IN_BV], vh, 16384, 256, 256);
    flash_kernel<<<Bsz*NHd*64,32,0,stream>>>(qh, kh, vh, ctx);
    gemm_kernel<0><<<g,256,0,stream>>>(ctx, (const float*)d_in[IN_WO],
        (const float*)d_in[IN_BO], oproj, 16384, 256, 256);
  }
  resln_kernel<<<(Bsz*LEN+255)/256,256,0,stream>>>(oproj, tf,
      (const float*)d_in[IN_LN_G], (const float*)d_in[IN_LN_B], fused);
  // 7) heads
  pool_kernel<<<(Bsz*HID+255)/256,256,0,stream>>>(fused, pooled);
  linear_kernel<<<(Bsz*128+255)/256,256,0,stream>>>(pooled, (const float*)d_in[IN_CLS1_W],
      (const float*)d_in[IN_CLS1_B], mid1, 256, 128, 1);
  linear_kernel<<<(Bsz*5+255)/256,256,0,stream>>>(mid1, (const float*)d_in[IN_CLS2_W],
      (const float*)d_in[IN_CLS2_B], out_cls, 128, 5, 0);
  // loc convs read fused transposed in-place (in_blc=1 for the first)
  conv1d_kernel<<<(Bsz*128*LEN+255)/256,256,0,stream>>>(fused, (const float*)d_in[IN_LOC1_W],
      (const float*)d_in[IN_LOC1_B], loc1b, 256, 128, LEN, 3, 1, 0, 128, 1, 1);
  conv1d_kernel<<<(Bsz*64*LEN+255)/256,256,0,stream>>>(loc1b, (const float*)d_in[IN_LOC2_W],
      (const float*)d_in[IN_LOC2_B], loc2b, 128, 64, LEN, 3, 1, 0, 64, 1, 0);
  conv1d_kernel<<<(Bsz*1*LEN+255)/256,256,0,stream>>>(loc2b, (const float*)d_in[IN_LOC3_W],
      (const float*)d_in[IN_LOC3_B], out_loc, 64, 1, LEN, 1, 0, 0, 1, 2, 0);
  linear_kernel<<<(Bsz*64+255)/256,256,0,stream>>>(pooled, (const float*)d_in[IN_SEV1_W],
      (const float*)d_in[IN_SEV1_B], mid2, 256, 64, 1);
  linear_kernel<<<(Bsz+255)/256,256,0,stream>>>(mid2, (const float*)d_in[IN_SEV2_W],
      (const float*)d_in[IN_SEV2_B], out_sev, 64, 1, 2);
  task_kernel<<<1,1,0,stream>>>((const float*)d_in[IN_TASK_W], out_tw);
}
